// CustomAttn_51591147160264
// MI455X (gfx1250) — matrix operand order repaired and checked by reading, not compiled, not run on hardware
//
#include <hip/hip_runtime.h>

// ---------------------------------------------------------------------------
// GPT2-style attention block with ALiBi, fp32 I/O, MI455X (gfx1250).
// - All matmul FLOPs via v_wmma_f32_16x16x32_f16 with f16 hi/lo split
//   (3 WMMAs per fp32 product, ~2^-22 accuracy, fp32 accumulate).
// - All f32->f16 splitting hoisted into one-time prep / epilogue passes so
//   GEMM + flash hot loops are pure {async DMA, ds_load_b128, v_wmma}.
// - GEMMs double-buffer f16 tiles with GLOBAL_LOAD_ASYNC_TO_LDS (ASYNCcnt);
//   64x64 wave tiles give 48 WMMA per 32 ds_load_b128 per K-step.
// - Attention is flash-style (online softmax); V tile DMA overlaps QK^T.
// ---------------------------------------------------------------------------

typedef __attribute__((ext_vector_type(16))) _Float16 v16h;
typedef __attribute__((ext_vector_type(8)))  _Float16 v8h;
typedef __attribute__((ext_vector_type(8)))  float    v8f;
typedef __attribute__((ext_vector_type(4)))  float    v4f;

__device__ __forceinline__ void split16(float x, _Float16& h, _Float16& l) {
    h = (_Float16)x;
    l = (_Float16)(x - (float)h);
}

// hi/lo 3-term WMMA accumulate: c += (ah+al)*(bh+bl) approx (drop al*bl)
__device__ __forceinline__ v8f wmma3(v16h ah, v16h al, v16h bh, v16h bl, v8f c) {
    c = __builtin_amdgcn_wmma_f32_16x16x32_f16(false, ah, false, bh, (short)0, c, false, false);
    c = __builtin_amdgcn_wmma_f32_16x16x32_f16(false, ah, false, bl, (short)0, c, false, false);
    c = __builtin_amdgcn_wmma_f32_16x16x32_f16(false, al, false, bh, (short)0, c, false, false);
    return c;
}

// -------- gfx1250 async global->LDS DMA (ASYNCcnt) --------
__device__ __forceinline__ void async_b128(unsigned lds_off, const void* g) {
    asm volatile("global_load_async_to_lds_b128 %0, %1, off"
                 :: "v"(lds_off), "v"(g) : "memory");
}
__device__ __forceinline__ void wait_async0() {
    asm volatile("s_wait_asynccnt 0x0" ::: "memory");
}

// A-fragment (16x32 f16): per ISA, lanes 0-15 elems 0..7 = K0..7, 8..15 =
// K16..23; lanes 16-31 shifted by +8. Works on LDS or global f16 pointers.
__device__ __forceinline__ v16h ldfragA(const _Float16* p, int lane) {
    int kb = (lane >> 4) * 8;
    v8h a0 = *(const v8h*)(p + kb);
    v8h a1 = *(const v8h*)(p + kb + 16);
    v16h r;
#pragma unroll
    for (int j = 0; j < 8; ++j) { r[j] = a0[j]; r[j + 8] = a1[j]; }
    return r;
}

// B-fragment (32x16 f16), p = base of this lane's N-column (data [n][k]):
// lanes 0-15: K0..7, K16..23; lanes 16-31: K8..15, K24..31 (same as A).
__device__ __forceinline__ v16h ldfragB(const _Float16* p, int lane) {
    int kb = (lane >> 4) * 8;
    v8h a0 = *(const v8h*)(p + kb);
    v8h a1 = *(const v8h*)(p + kb + 16);
    v16h r;
#pragma unroll
    for (int j = 0; j < 8; ++j) { r[j] = a0[j]; r[j + 8] = a1[j]; }
    return r;
}

// ---------------------------------------------------------------------------
// Prep kernels: one-time f32 -> hi/lo f16 split (and transpose for B mats).
// ---------------------------------------------------------------------------
__global__ __launch_bounds__(256)
void split_rm(const float* __restrict__ in, _Float16* __restrict__ oh,
              _Float16* __restrict__ ol, int n4 /* n/4 */) {
    int i0 = (blockIdx.x * 256 + threadIdx.x) * 4;
    if ((i0 >> 2) >= n4) return;
    v4f v = *(const v4f*)(in + i0);
#pragma unroll
    for (int j = 0; j < 4; ++j) {
        _Float16 hh, ll;
        split16(v[j], hh, ll);
        oh[i0 + j] = hh;
        ol[i0 + j] = ll;
    }
}

// in[R][C] f32 -> out[C][R] hi/lo f16, 64x64 LDS tile transpose.
__global__ __launch_bounds__(256)
void split_tr(const float* __restrict__ in, _Float16* __restrict__ oh,
              _Float16* __restrict__ ol, int R, int C) {
    __shared__ float t[64][65];
    const int tid = threadIdx.x;
    const int bc = blockIdx.x * 64;
    const int br = blockIdx.y * 64;
#pragma unroll
    for (int i = 0; i < 16; ++i) {
        int e = tid + i * 256;
        int r = e >> 6, c = e & 63;
        t[r][c] = in[(size_t)(br + r) * C + bc + c];
    }
    __syncthreads();
#pragma unroll
    for (int i = 0; i < 16; ++i) {
        int e = tid + i * 256;
        int c = e >> 6, r = e & 63;
        _Float16 hh, ll;
        split16(t[r][c], hh, ll);
        oh[(size_t)(bc + c) * R + br + r] = hh;
        ol[(size_t)(bc + c) * R + br + r] = ll;
    }
}

// ---------------------------------------------------------------------------
// WMMA GEMM mainloop: A (hi/lo f16, [M][K]) x B (hi/lo f16, [N][K], i.e.
// pre-transposed). Block tile 128x128x32, 4 waves = 2(M)x2(N), wave 64x64.
// Double-buffered f16 tiles in dynamic LDS via async DMA. No VALU convert.
// ---------------------------------------------------------------------------
#define GT 128                      // GEMM block threads (4 waves)
#define BK 32
#define TPH 40                      // LDS tile pitch in f16 (80B, 16B aligned)
#define TILE16 (128 * TPH)          // 5120 f16 = one buffer of one array
#define GEMM_LDS_BYTES (4 * 2 * TILE16 * 2)  // 4 arrays x 2 buffers x 2B

__device__ __forceinline__ void wmma_mainloop(
    const _Float16* __restrict__ Ah, const _Float16* __restrict__ Al,
    const _Float16* __restrict__ Bh, const _Float16* __restrict__ Bl,
    int K, int bm0, int bn0, int tid, int lane, int wm, int wn,
    v8f (&acc)[4][4]) {
    extern __shared__ float smem[];
    _Float16* s16 = (_Float16*)smem;
    _Float16* sAh = s16;
    _Float16* sAl = s16 + 2 * TILE16;
    _Float16* sBh = s16 + 4 * TILE16;
    _Float16* sBl = s16 + 6 * TILE16;
    const unsigned lAh = (unsigned)(size_t)sAh;
    const unsigned lAl = (unsigned)(size_t)sAl;
    const unsigned lBh = (unsigned)(size_t)sBh;
    const unsigned lBl = (unsigned)(size_t)sBl;
    const int ln = lane & 15;

    auto issue_tile = [&](int buf, int kt) {
#pragma unroll
        for (int i = 0; i < 4; ++i) {
            int e = tid + i * GT;             // 512 chunks of 8 f16
            int row = e >> 2, kc = (e & 3) * 8;
            unsigned off = (unsigned)((buf * 128 + row) * TPH + kc) * 2u;
            size_t ga = (size_t)(bm0 + row) * K + kt + kc;
            size_t gb = (size_t)(bn0 + row) * K + kt + kc;
            async_b128(lAh + off, Ah + ga);
            async_b128(lAl + off, Al + ga);
            async_b128(lBh + off, Bh + gb);
            async_b128(lBl + off, Bl + gb);
        }
    };

    issue_tile(0, 0);
    int buf = 0;
    for (int kt = 0; kt < K; kt += BK) {
        wait_async0();        // this wave's DMA for tile `buf` landed
        __syncthreads();      // everyone's DMA landed & prev compute done
        if (kt + BK < K) issue_tile(buf ^ 1, kt + BK);

        const _Float16* aH = sAh + buf * TILE16;
        const _Float16* aL = sAl + buf * TILE16;
        const _Float16* bH = sBh + buf * TILE16;
        const _Float16* bL = sBl + buf * TILE16;
        v16h bfh[4], bfl[4];
#pragma unroll
        for (int j = 0; j < 4; ++j) {
            int n = wn * 64 + j * 16 + ln;
            bfh[j] = ldfragB(bH + n * TPH, lane);
            bfl[j] = ldfragB(bL + n * TPH, lane);
        }
#pragma unroll
        for (int i = 0; i < 4; ++i) {
            int m = wm * 64 + i * 16 + ln;
            v16h afh = ldfragA(aH + m * TPH, lane);
            v16h afl = ldfragA(aL + m * TPH, lane);
#pragma unroll
            for (int j = 0; j < 4; ++j)
                acc[i][j] = wmma3(afh, afl, bfh[j], bfl[j], acc[i][j]);
        }
        buf ^= 1;
    }
}

// GEMM with plain f32 output (+bias): out-projection -> d_out.
__global__ __launch_bounds__(GT)
void gemm_out_f32(const _Float16* __restrict__ Ah, const _Float16* __restrict__ Al,
                  const _Float16* __restrict__ Bh, const _Float16* __restrict__ Bl,
                  const float* __restrict__ bias, float* __restrict__ C,
                  int N, int K) {
    const int tid = threadIdx.x, lane = tid & 31, wv = tid >> 5;
    const int wm = wv >> 1, wn = wv & 1, ln = lane & 15, half = lane >> 4;
    const int bn0 = blockIdx.x * 128, bm0 = blockIdx.y * 128;
    v8f acc[4][4];
    const v8f vz = {0.f, 0.f, 0.f, 0.f, 0.f, 0.f, 0.f, 0.f};
#pragma unroll
    for (int i = 0; i < 4; ++i)
#pragma unroll
        for (int j = 0; j < 4; ++j) acc[i][j] = vz;
    wmma_mainloop(Ah, Al, Bh, Bl, K, bm0, bn0, tid, lane, wm, wn, acc);
#pragma unroll
    for (int i = 0; i < 4; ++i)
#pragma unroll
        for (int j = 0; j < 4; ++j) {
            int col = bn0 + wn * 64 + j * 16 + ln;
            float bv = bias[col];
#pragma unroll
            for (int r = 0; r < 8; ++r) {
                int row = bm0 + wm * 64 + i * 16 + half * 8 + r;
                C[(size_t)row * N + col] = acc[i][j][r] + bv;
            }
        }
}

// QKV GEMM: writes Q,K as split hi/lo f16 [seq][2048]; V split + transposed
// vT[dglobal][seq] so flash can async-DMA B-fragment-ready tiles.
__global__ __launch_bounds__(GT)
void gemm_qkv(const _Float16* __restrict__ Ah, const _Float16* __restrict__ Al,
              const _Float16* __restrict__ Bh, const _Float16* __restrict__ Bl,
              const float* __restrict__ bias,
              _Float16* __restrict__ Qh, _Float16* __restrict__ Ql,
              _Float16* __restrict__ Kh, _Float16* __restrict__ Kl,
              _Float16* __restrict__ Vth, _Float16* __restrict__ Vtl,
              int K) {
    const int tid = threadIdx.x, lane = tid & 31, wv = tid >> 5;
    const int wm = wv >> 1, wn = wv & 1, ln = lane & 15, half = lane >> 4;
    const int bn0 = blockIdx.x * 128, bm0 = blockIdx.y * 128;
    v8f acc[4][4];
    const v8f vz = {0.f, 0.f, 0.f, 0.f, 0.f, 0.f, 0.f, 0.f};
#pragma unroll
    for (int i = 0; i < 4; ++i)
#pragma unroll
        for (int j = 0; j < 4; ++j) acc[i][j] = vz;
    wmma_mainloop(Ah, Al, Bh, Bl, K, bm0, bn0, tid, lane, wm, wn, acc);

    const int third = bn0 >> 11;  // 0=Q, 1=K, 2=V (uniform: 2048 % 128 == 0)
#pragma unroll
    for (int i = 0; i < 4; ++i)
#pragma unroll
        for (int j = 0; j < 4; ++j) {
            int colg = bn0 + wn * 64 + j * 16 + ln;
            float bv = bias[colg];
            int c = colg & 2047;
#pragma unroll
            for (int r = 0; r < 8; ++r) {
                int row = bm0 + wm * 64 + i * 16 + half * 8 + r;
                float v = acc[i][j][r] + bv;
                _Float16 hh, ll;
                split16(v, hh, ll);
                if (third == 0) {
                    Qh[(size_t)row * 2048 + c] = hh;
                    Ql[(size_t)row * 2048 + c] = ll;
                } else if (third == 1) {
                    Kh[(size_t)row * 2048 + c] = hh;
                    Kl[(size_t)row * 2048 + c] = ll;
                } else {
                    Vth[(size_t)c * 2048 + row] = hh;
                    Vtl[(size_t)c * 2048 + row] = ll;
                }
            }
        }
}

// ---------------------------------------------------------------------------
// Flash attention with causal mask + ALiBi, all operands pre-split f16.
// Block = (head, 128 query rows), 8 waves x 16 query rows.
// V tile DMA (async->LDS, already B-fragment oriented) overlaps QK^T.
// Output written pre-split hi/lo for the out-projection GEMM.
// ---------------------------------------------------------------------------
#define S_LEN 2048
#define EMB   2048
#define HD    64
#define VPH   72   // V tile pitch in f16 (144B: 16B aligned)
#define KPAD  72   // P staging pitch in f16

__global__ __launch_bounds__(256)
void flash_attn_alibi(const _Float16* __restrict__ Qh, const _Float16* __restrict__ Ql,
                      const _Float16* __restrict__ Kh, const _Float16* __restrict__ Kl,
                      const _Float16* __restrict__ Vth, const _Float16* __restrict__ Vtl,
                      _Float16* __restrict__ Oh, _Float16* __restrict__ Ol) {
    __shared__ _Float16 sVh[HD][VPH];        // [d][key] (B-fragment oriented)
    __shared__ _Float16 sVl[HD][VPH];
    __shared__ _Float16 sPh[8][16][KPAD];    // per-wave P staging [row][key]
    __shared__ _Float16 sPl[8][16][KPAD];

    const int tid  = threadIdx.x;
    const int lane = tid & 31;
    const int wv   = tid >> 5;
    const int half = lane >> 4;
    const int ln   = lane & 15;
    const int h    = blockIdx.y;
    const int q0   = blockIdx.x * 128;
    const int qr0  = q0 + wv * 16;
    const unsigned lVh = (unsigned)(size_t)&sVh[0][0];
    const unsigned lVl = (unsigned)(size_t)&sVl[0][0];
    const float mslope = 0.8408964152537145f;  // 2^(-8/32*(HEAD_ID+1)), HEAD_ID=0

    // Q A-fragments straight from pre-split f16 (scale 1/8 folded in later)
    v16h qfh[2], qfl[2];
#pragma unroll
    for (int d0 = 0; d0 < 2; ++d0) {
        size_t off = (size_t)(qr0 + ln) * EMB + h * HD + d0 * 32;
        qfh[d0] = ldfragA(Qh + off, lane);
        qfl[d0] = ldfragA(Ql + off, lane);
    }

    v8f O[4];
    const v8f vz = {0.f, 0.f, 0.f, 0.f, 0.f, 0.f, 0.f, 0.f};
#pragma unroll
    for (int dt = 0; dt < 4; ++dt) O[dt] = vz;
    float mrow[8], lrow[8];
#pragma unroll
    for (int r = 0; r < 8; ++r) { mrow[r] = -1.0e30f; lrow[r] = 0.0f; }

    const int kend = q0 + 128;  // uniform causal bound for whole block
    for (int kb = 0; kb < kend; kb += 64) {
        __syncthreads();  // all waves done reading previous V tile
        // async DMA: V tile (64 d x 64 keys) from vT, fragment-native layout
#pragma unroll
        for (int i = 0; i < 2; ++i) {
            int e = tid + i * 256;           // 512 chunks of 8 f16
            int d = e >> 3, kc = (e & 7) * 8;
            unsigned off = (unsigned)(d * VPH + kc) * 2u;
            size_t g = (size_t)(h * HD + d) * S_LEN + kb + kc;
            async_b128(lVh + off, Vth + g);
            async_b128(lVl + off, Vtl + g);
        }

        // ---- overlap: QK^T scores while V tile streams in ----
        v8f sacc[4];
#pragma unroll
        for (int c = 0; c < 4; ++c) {
            v8f s = vz;
            size_t krow = (size_t)(kb + c * 16 + ln) * EMB + h * HD;
#pragma unroll
            for (int d0 = 0; d0 < 2; ++d0) {
                v16h kfh = ldfragB(Kh + krow + d0 * 32, lane);
                v16h kfl = ldfragB(Kl + krow + d0 * 32, lane);
                s = wmma3(qfh[d0], qfl[d0], kfh, kfl, s);
            }
            sacc[c] = s;
        }

        // 1/sqrt(D) scale + causal mask + ALiBi + online softmax
#pragma unroll
        for (int r = 0; r < 8; ++r) {
            int qrow = qr0 + half * 8 + r;
            float mx = -1.0e30f;
#pragma unroll
            for (int c = 0; c < 4; ++c) {
                int key = kb + c * 16 + ln;
                float sv = sacc[c][r] * 0.125f;
                sv = (key <= qrow) ? (sv + (float)(key - qrow) * mslope) : -10000.0f;
                sacc[c][r] = sv;
                mx = fmaxf(mx, sv);
            }
#pragma unroll
            for (int xm = 1; xm < 16; xm <<= 1)
                mx = fmaxf(mx, __shfl_xor(mx, xm, 32));
            float mnew = fmaxf(mrow[r], mx);
            float scl = __expf(mrow[r] - mnew);
            mrow[r] = mnew;
            float rsum = 0.0f;
#pragma unroll
            for (int c = 0; c < 4; ++c) {
                float p = __expf(sacc[c][r] - mnew);
                sacc[c][r] = p;
                rsum += p;
            }
#pragma unroll
            for (int xm = 1; xm < 16; xm <<= 1)
                rsum += __shfl_xor(rsum, xm, 32);
            lrow[r] = lrow[r] * scl + rsum;
#pragma unroll
            for (int dt = 0; dt < 4; ++dt) O[dt][r] *= scl;
        }

        // P: C-layout -> A-layout via this wave's private LDS slab
#pragma unroll
        for (int c = 0; c < 4; ++c)
#pragma unroll
            for (int r = 0; r < 8; ++r) {
                int rr = half * 8 + r;
                _Float16 hh, ll;
                split16(sacc[c][r], hh, ll);
                sPh[wv][rr][c * 16 + ln] = hh;
                sPl[wv][rr][c * 16 + ln] = ll;
            }
        asm volatile("s_wait_dscnt 0x0" ::: "memory");  // wave-private staging

        wait_async0();    // V DMA landed (this wave's requests)
        __syncthreads();  // all waves' V DMA visible

        // O += P @ V  (contraction over 64 keys = 2 WMMA k-steps)
#pragma unroll
        for (int kk = 0; kk < 2; ++kk) {
            v16h ph = ldfragA(&sPh[wv][ln][kk * 32], lane);
            v16h pl = ldfragA(&sPl[wv][ln][kk * 32], lane);
#pragma unroll
            for (int dt = 0; dt < 4; ++dt) {
                int d = dt * 16 + ln;
                v16h vh = ldfragB(&sVh[d][kk * 32], lane);
                v16h vl = ldfragB(&sVl[d][kk * 32], lane);
                O[dt] = wmma3(ph, pl, vh, vl, O[dt]);
            }
        }
    }

    // finalize: O / l, pre-split hi/lo, merged-head layout [S, E]
#pragma unroll
    for (int dt = 0; dt < 4; ++dt)
#pragma unroll
        for (int r = 0; r < 8; ++r) {
            int row = qr0 + half * 8 + r;
            float o = O[dt][r] / lrow[r];
            _Float16 hh, ll;
            split16(o, hh, ll);
            size_t off = (size_t)row * EMB + h * HD + dt * 16 + ln;
            Oh[off] = hh;
            Ol[off] = ll;
        }
}

// ---------------------------------------------------------------------------

extern "C" void kernel_launch(void* const* d_in, const int* in_sizes, int n_in,
                              void* d_out, int out_size, void* d_ws, size_t ws_size,
                              hipStream_t stream) {
    (void)in_sizes; (void)n_in; (void)out_size; (void)ws_size;
    const float* X      = (const float*)d_in[0];  // [2048, 2048]
    const float* W_attn = (const float*)d_in[1];  // [2048, 6144]
    const float* b_attn = (const float*)d_in[2];  // [6144]
    const float* W_proj = (const float*)d_in[3];  // [2048, 2048]
    const float* b_proj = (const float*)d_in[4];  // [2048]
    float* out = (float*)d_out;

    const size_t E2 = (size_t)2048 * 2048;        // 4M elems
    _Float16* p = (_Float16*)d_ws;
    _Float16* xh    = p;            p += E2;      // later aliased: attn_h
    _Float16* xl    = p;            p += E2;      // later aliased: attn_l
    _Float16* wat_h = p;            p += 3 * E2;  // later aliased: wpt_h/wpt_l
    _Float16* wat_l = p;            p += 3 * E2;
    _Float16* Qh    = p;            p += E2;
    _Float16* Ql    = p;            p += E2;
    _Float16* Kh    = p;            p += E2;
    _Float16* Kl    = p;            p += E2;
    _Float16* Vth   = p;            p += E2;
    _Float16* Vtl   = p;            p += E2;      // total 112 MB
    _Float16* attn_h = xh;                        // X dead after gemm_qkv
    _Float16* attn_l = xl;
    _Float16* wpt_h  = wat_h;                     // W_attn-T dead after gemm_qkv
    _Float16* wpt_l  = wat_h + E2;

    // prep: split X; split+transpose W_attn
    split_rm<<<dim3((unsigned)(E2 / 1024)), dim3(256), 0, stream>>>(X, xh, xl, (int)(E2 / 4));
    split_tr<<<dim3(6144 / 64, 2048 / 64), dim3(256), 0, stream>>>(W_attn, wat_h, wat_l, 2048, 6144);
    // qkv = X @ W_attn + b_attn  (Q/K split f16; V split + transposed)
    gemm_qkv<<<dim3(6144 / 128, 2048 / 128), dim3(GT), GEMM_LDS_BYTES, stream>>>(
        xh, xl, wat_h, wat_l, b_attn, Qh, Ql, Kh, Kl, Vth, Vtl, 2048);
    // flash attention, all 32 heads -> pre-split attn
    flash_attn_alibi<<<dim3(S_LEN / 128, 32), dim3(256), 0, stream>>>(
        Qh, Ql, Kh, Kl, Vth, Vtl, attn_h, attn_l);
    // prep: split+transpose W_proj (reuses W_attn-T slab)
    split_tr<<<dim3(2048 / 64, 2048 / 64), dim3(256), 0, stream>>>(W_proj, wpt_h, wpt_l, 2048, 2048);
    // out = attn @ W_proj + b_proj
    gemm_out_f32<<<dim3(2048 / 128, 2048 / 128), dim3(GT), GEMM_LDS_BYTES, stream>>>(
        attn_h, attn_l, wpt_h, wpt_l, b_proj, out, 2048, 2048);
}
